// Cross_Attention_90348932038964
// MI455X (gfx1250) — compile-verified, matrix operand order checked by reading, not run on hardware
//
#include <hip/hip_runtime.h>

// ---------------------------------------------------------------------------
// CDNA5 (gfx1250) cross-attention: all matrix math on v_wmma_f32_16x16x32_bf16,
// bf16 operands / f32 accumulation, global_prefetch on the streamed operands.
// ---------------------------------------------------------------------------

typedef __attribute__((ext_vector_type(16))) __bf16 v16bf;
typedef __attribute__((ext_vector_type(8)))  float  v8f;

union ABFrag {
    v16bf v;
    unsigned int u[8];
};

// Per-lane K offset for 16-bit A/B fragments (16x32 / 32x16), wave32.
// lane half 0 (lanes 0-15): VGPR0..3 -> K 0..7,  VGPR4..7 -> K 16..23
// lane half 1 (lanes16-31): VGPR0..3 -> K 8..15, VGPR4..7 -> K 24..31
__device__ __forceinline__ int koff(int i, int kh) {
    return ((i < 4) ? 0 : 16) + kh * 8 + 2 * (i & 3);
}

// f32 -> bf16 via the native conversion (RNE); backend emits the hardware cvt.
__device__ __forceinline__ unsigned short f2bf(float x) {
    union { __bf16 h; unsigned short s; } cv;
    cv.h = (__bf16)x;
    return cv.s;
}

__device__ __forceinline__ unsigned int f2bf2(float lo, float hi) {
    return (unsigned int)f2bf(lo) | ((unsigned int)f2bf(hi) << 16);
}

// ---------------------------------------------------------------------------
// Projection GEMM: Out(bf16) = X[8192x1024](f32) * W[1024x1024](f32) + b
// Block tile 128x128, K-slab 64 (16 WMMA per barrier pair), 8 waves each own
// a 32x64 sub-tile. fp32->bf16 conversion fused into LDS staging.
// transOut == 0: Out[row*1024 + col]                  (row-major, Q/K)
// transOut != 0: Out[b][col][n], n = row % 2048       (col-major per batch, V)
//                -> lane's 8 accum rows become one contiguous b128 store.
// ---------------------------------------------------------------------------
__global__ __launch_bounds__(256) void proj_gemm_bf16(
    const float* __restrict__ X, const float* __restrict__ W,
    const float* __restrict__ bias, unsigned short* __restrict__ Out,
    int transOut)
{
    __shared__ unsigned short lds_a[128 * 64];   // X tile, row-major [m][k]
    __shared__ unsigned short lds_b[128 * 64];   // W tile, transposed [n][k]

    const int t    = threadIdx.x;
    const int lane = t & 31;
    const int wave = t >> 5;

    const int blockM = blockIdx.x * 128;
    const int blockN = blockIdx.y * 128;
    const int waveM  = (wave & 3) * 32;
    const int waveN  = (wave >> 2) * 64;

    const int m16 = lane & 15;
    const int kh  = lane >> 4;

    v8f zero = {};
    v8f acc[2][4];
    #pragma unroll
    for (int mt = 0; mt < 2; ++mt)
        #pragma unroll
        for (int nt = 0; nt < 4; ++nt) acc[mt][nt] = zero;

    for (int k0 = 0; k0 < 1024; k0 += 64) {
        // Stage X tile [128][64] as bf16, row-major (K pairs contiguous)
        {
            const int row   = t >> 1;
            const int cbase = (t & 1) * 32;
            const float* src = X + (size_t)(blockM + row) * 1024 + k0 + cbase;
            unsigned short* dst = &lds_a[row * 64 + cbase];
            __builtin_prefetch(src + 64, 0, 3);        // next k-slab, same row
            #pragma unroll
            for (int i = 0; i < 32; i += 2)
                *(unsigned int*)&dst[i] = f2bf2(src[i], src[i + 1]);
        }
        // Stage W tile [64][128] transposed -> [n][k] (K pairs contiguous)
        {
            const int kk    = t >> 2;
            const int nbase = (t & 3) * 32;
            const float* src = W + (size_t)(k0 + kk) * 1024 + blockN + nbase;
            __builtin_prefetch(src + (size_t)64 * 1024, 0, 3); // next k-slab rows
            #pragma unroll
            for (int i = 0; i < 32; ++i) lds_b[(nbase + i) * 64 + kk] = f2bf(src[i]);
        }
        __syncthreads();

        #pragma unroll
        for (int kc = 0; kc < 64; kc += 32) {
            ABFrag a[2], b[4];
            #pragma unroll
            for (int mt = 0; mt < 2; ++mt) {
                const int row = waveM + mt * 16 + m16;
                #pragma unroll
                for (int i = 0; i < 8; ++i)
                    a[mt].u[i] = *(const unsigned int*)&lds_a[row * 64 + kc + koff(i, kh)];
            }
            #pragma unroll
            for (int nt = 0; nt < 4; ++nt) {
                const int col = waveN + nt * 16 + m16;
                #pragma unroll
                for (int i = 0; i < 8; ++i)
                    b[nt].u[i] = *(const unsigned int*)&lds_b[col * 64 + kc + koff(i, kh)];
            }
            #pragma unroll
            for (int mt = 0; mt < 2; ++mt)
                #pragma unroll
                for (int nt = 0; nt < 4; ++nt)
                    acc[mt][nt] = __builtin_amdgcn_wmma_f32_16x16x32_bf16(
                        false, a[mt].v, false, b[nt].v, (short)0, acc[mt][nt],
                        false, false);
        }
        __syncthreads();
    }

    // Epilogue. C/D layout: lane n = lane&15, VGPR r -> m = r + 8*(lane>>4)
    #pragma unroll
    for (int nt = 0; nt < 4; ++nt) {
        const int col = blockN + waveN + nt * 16 + m16;
        const float bv = bias[col];
        #pragma unroll
        for (int mt = 0; mt < 2; ++mt) {
            const int rowBase = blockM + waveM + mt * 16 + kh * 8;
            if (transOut) {
                // Vt[b][col][n]: 8 consecutive n -> one 16B store
                const size_t idx = (size_t)(rowBase >> 11) * (1024u * 2048u)
                                 + (size_t)col * 2048u + (rowBase & 2047);
                uint4 pk;
                pk.x = f2bf2(acc[mt][nt][0] + bv, acc[mt][nt][1] + bv);
                pk.y = f2bf2(acc[mt][nt][2] + bv, acc[mt][nt][3] + bv);
                pk.z = f2bf2(acc[mt][nt][4] + bv, acc[mt][nt][5] + bv);
                pk.w = f2bf2(acc[mt][nt][6] + bv, acc[mt][nt][7] + bv);
                *(uint4*)&Out[idx] = pk;
            } else {
                #pragma unroll
                for (int r = 0; r < 8; ++r)
                    Out[(size_t)(rowBase + r) * 1024 + col] = f2bf(acc[mt][nt][r] + bv);
            }
        }
    }
}

// ---------------------------------------------------------------------------
// Attention: one block (8 waves) per (batch, 16-query tile).
// Phase 0: stage Q tile [16][1024] bf16 -> LDS (coalesced uint4)
// Phase 1: S[16][2048] = scale * Q Kt; keys split across waves; A-frags from
//          LDS (ds_load_b128), B-frags directly from global bf16 K (row-major
//          [key][c] == B-fragment layout); prefetch next key tile's rows.
// Phase 2: row softmax in LDS (f32), re-quantize P to bf16 in LDS
// Phase 3: O = P V using transposed Vt[b][c][n]: B-frags are aligned 32-bit
//          global loads; each wave owns a 128-wide column slice.
// LDS: S 128K + P 64K + Q 32K + scratch ~= 225 KB of the WGP's 320 KB.
// ---------------------------------------------------------------------------
__global__ __launch_bounds__(256) void cross_attn_bf16(
    const unsigned short* __restrict__ Q, const unsigned short* __restrict__ K,
    const unsigned short* __restrict__ Vt, float* __restrict__ Out, float scale)
{
    extern __shared__ char smem[];
    float*          S       = (float*)smem;                               // 16*2048 f32
    unsigned short* P       = (unsigned short*)(smem + 131072);           // 16*2048 bf16
    unsigned short* Qs      = (unsigned short*)(smem + 131072 + 65536);   // 16*1024 bf16
    float*          red     = (float*)(smem + 131072 + 65536 + 32768);    // 16*16
    float*          rowstat = red + 256;                                  // 16

    const int t    = threadIdx.x;
    const int lane = t & 31;
    const int wave = t >> 5;
    const int m16  = lane & 15;
    const int kh   = lane >> 4;

    const int b  = blockIdx.x >> 7;     // 128 query tiles per batch
    const int qt = blockIdx.x & 127;

    const unsigned short* Qb  = Q  + ((size_t)b * 2048 + qt * 16) * 1024;
    const unsigned short* Kb  = K  + (size_t)b * 2048 * 1024;
    const unsigned short* Vtb = Vt + (size_t)b * 1024 * 2048;

    // ---- Phase 0: stage Q tile into LDS ----
    {
        uint4*       qd = (uint4*)Qs;
        const uint4* qs = (const uint4*)Qb;
        #pragma unroll
        for (int j = 0; j < 8; ++j) qd[t + 256 * j] = qs[t + 256 * j];
    }
    __syncthreads();

    // ---- Phase 1: scores ----
    for (int kt = 0; kt < 16; ++kt) {
        const int key0 = wave * 256 + kt * 16;
        const unsigned short* krow = &Kb[(size_t)(key0 + m16) * 1024];
        v8f acc = {};
        for (int c0 = 0; c0 < 1024; c0 += 32) {
            __builtin_prefetch(krow + 16 * 1024 + c0, 0, 3);  // next key tile
            ABFrag a, bb;
            #pragma unroll
            for (int i = 0; i < 8; ++i) {
                const int c = c0 + koff(i, kh);
                a.u[i]  = *(const unsigned int*)&Qs[m16 * 1024 + c];
                bb.u[i] = *(const unsigned int*)&krow[c];
            }
            acc = __builtin_amdgcn_wmma_f32_16x16x32_bf16(
                false, a.v, false, bb.v, (short)0, acc, false, false);
        }
        #pragma unroll
        for (int r = 0; r < 8; ++r) {
            const int m = kh * 8 + r;
            S[m * 2048 + key0 + m16] = acc[r] * scale;
        }
    }
    __syncthreads();

    // ---- Phase 2: softmax over each of 16 rows (2048 wide) ----
    const int row = t >> 4;    // 0..15
    const int sub = t & 15;    // 16 threads per row, stride-16 columns

    float pm = -3.0e38f;
    for (int k = sub; k < 2048; k += 16) pm = fmaxf(pm, S[row * 2048 + k]);
    red[row * 16 + sub] = pm;
    __syncthreads();
    if (sub == 0) {
        float m = red[row * 16];
        for (int i = 1; i < 16; ++i) m = fmaxf(m, red[row * 16 + i]);
        rowstat[row] = m;
    }
    __syncthreads();
    const float rmax = rowstat[row];
    float ps = 0.f;
    for (int k = sub; k < 2048; k += 16) {
        const float e = __expf(S[row * 2048 + k] - rmax);
        S[row * 2048 + k] = e;
        ps += e;
    }
    red[row * 16 + sub] = ps;
    __syncthreads();
    if (sub == 0) {
        float s = 0.f;
        for (int i = 0; i < 16; ++i) s += red[row * 16 + i];
        rowstat[row] = 1.0f / s;
    }
    __syncthreads();
    const float rinv = rowstat[row];
    for (int k = sub; k < 2048; k += 16)
        P[row * 2048 + k] = f2bf(S[row * 2048 + k] * rinv);
    __syncthreads();

    // ---- Phase 3: O = P V, wave owns cols [wave*128, wave*128+128) ----
    v8f zero = {};
    v8f o[8];
    #pragma unroll
    for (int nt = 0; nt < 8; ++nt) o[nt] = zero;

    const int colBase = wave * 128;
    for (int k0 = 0; k0 < 2048; k0 += 32) {
        ABFrag a;
        #pragma unroll
        for (int i = 0; i < 8; ++i)
            a.u[i] = *(const unsigned int*)&P[m16 * 2048 + k0 + koff(i, kh)];
        #pragma unroll
        for (int nt = 0; nt < 8; ++nt) {
            const int col = colBase + nt * 16 + m16;
            const unsigned short* vcol = &Vtb[(size_t)col * 2048];
            __builtin_prefetch(vcol + k0 + 512, 0, 3);
            ABFrag bb;
            #pragma unroll
            for (int i = 0; i < 8; ++i)
                bb.u[i] = *(const unsigned int*)&vcol[k0 + koff(i, kh)];
            o[nt] = __builtin_amdgcn_wmma_f32_16x16x32_bf16(
                false, a.v, false, bb.v, (short)0, o[nt], false, false);
        }
    }

    float* Ob = Out + ((size_t)b * 2048 + qt * 16) * 1024;
    #pragma unroll
    for (int nt = 0; nt < 8; ++nt) {
        const int col = colBase + nt * 16 + m16;
        #pragma unroll
        for (int r = 0; r < 8; ++r) {
            const int m = kh * 8 + r;
            Ob[(size_t)m * 1024 + col] = o[nt][r];
        }
    }
}

// ---------------------------------------------------------------------------
// Launch
// ---------------------------------------------------------------------------
extern "C" void kernel_launch(void* const* d_in, const int* in_sizes, int n_in,
                              void* d_out, int out_size, void* d_ws, size_t ws_size,
                              hipStream_t stream) {
    (void)in_sizes; (void)n_in; (void)out_size; (void)ws_size;

    const float* event_f = (const float*)d_in[0];
    const float* img_f   = (const float*)d_in[1];
    const float* Wq      = (const float*)d_in[2];
    const float* bq      = (const float*)d_in[3];
    const float* Wk      = (const float*)d_in[4];
    const float* bk      = (const float*)d_in[5];
    const float* Wv      = (const float*)d_in[6];
    const float* bv      = (const float*)d_in[7];

    // Workspace: bf16 Q[8192][1024], K[8192][1024], Vt[4][1024][2048] -> 48 MB
    unsigned short* Qbf = (unsigned short*)d_ws;
    unsigned short* Kbf = Qbf + (size_t)8192 * 1024;
    unsigned short* Vtb = Kbf + (size_t)8192 * 1024;

    dim3 gridP(64, 8);   // M/128, N/128
    proj_gemm_bf16<<<gridP, 256, 0, stream>>>(event_f, Wq, bq, Qbf, 0);
    proj_gemm_bf16<<<gridP, 256, 0, stream>>>(img_f,   Wk, bk, Kbf, 0);
    proj_gemm_bf16<<<gridP, 256, 0, stream>>>(img_f,   Wv, bv, Vtb, 1);

    const size_t shmem = 131072          // S (f32)
                       + 65536           // P (bf16)
                       + 32768           // Q tile (bf16)
                       + (256 + 16) * 4; // reduction scratch
    const float scale = 0.03125f;        // 1/sqrt(1024)
    cross_attn_bf16<<<dim3(512), 256, shmem, stream>>>(Qbf, Kbf, Vtb,
                                                       (float*)d_out, scale);
}
